// CausalSelfAttention_8796093022941
// MI455X (gfx1250) — matrix operand order repaired and checked by reading, not compiled, not run on hardware
//
#include <hip/hip_runtime.h>
#include <hip/hip_bf16.h>

typedef __attribute__((ext_vector_type(16))) __bf16 v16bf;
typedef __attribute__((ext_vector_type(8)))  __bf16 v8bf;
typedef __attribute__((ext_vector_type(8)))  float  v8f;

#define B_  2
#define S_  2048
#define D_  2048
#define H_  16
#define KV_ 4
#define HD_ 128

static constexpr float SCALE_ = 0.08838834764831845f; // 128^-0.5

union V16U { v16bf v; v8bf h8[2]; };

__device__ __forceinline__ V16U ldfragA(const __bf16* arow, int kc, int ka0) {
  V16U a;
  a.h8[0] = *(const v8bf*)(arow + kc + ka0);
  a.h8[1] = *(const v8bf*)(arow + kc + ka0 + 16);
  return a;
}
__device__ __forceinline__ V16U ldfragB(const __bf16* bp) {
  V16U b;
  b.h8[0] = *(const v8bf*)(bp);
  b.h8[1] = *(const v8bf*)(bp + 16);
  return b;
}

// ---------------------------------------------------------------- casts
__global__ __launch_bounds__(256) void k_cast_bf16(const float* __restrict__ in,
                                                   __bf16* __restrict__ out, int n) {
  int i = blockIdx.x * blockDim.x + threadIdx.x;
  if (i < n) out[i] = (__bf16)in[i];
}

// in: f32 [K x N] row-major  ->  out: bf16 [N x K] row-major (transposed)
__global__ __launch_bounds__(256) void k_transpose_bf16(const float* __restrict__ in,
                                                        __bf16* __restrict__ out,
                                                        int K, int N) {
  int i = blockIdx.x * blockDim.x + threadIdx.x;
  if (i < K * N) {
    int k = i / N, n = i % N;
    out[(size_t)n * K + k] = (__bf16)in[i];
  }
}

// ---------------------------------------------------------------- GEMM
// C[M x N] f32 = A[M x K] bf16 (row-major) * B, with B given as BT[N x K] bf16.
// One wave computes a 32(M) x 64(N) tile: 8 accumulators, K stepped by 32,
// double-buffered fragment loads (requires K % 64 == 0).
__global__ __launch_bounds__(256) void k_gemm_bf16(const __bf16* __restrict__ A,
                                                   const __bf16* __restrict__ BT,
                                                   float* __restrict__ C,
                                                   int M, int N, int K) {
  const int lane = threadIdx.x & 31;
  const int wave = blockIdx.x * 8 + (threadIdx.x >> 5);
  const int ntiles = N >> 6;
  const int mt = wave / ntiles;
  const int nt = wave % ntiles;
  if (mt >= (M >> 5)) return;
  const int m0 = mt << 5, n0 = nt << 6;
  const int l16 = lane & 15;
  const int hi  = lane >> 4;
  const int ka0 = hi ? 8 : 0;    // A-frag K offset (ISA 16-bit A 16x32 layout)
  const int kb0 = hi ? 8 : 0;    // B-frag K offset (same K layout as A-frag)

  const __bf16* arow0 = A + (size_t)(m0 + l16) * K;
  const __bf16* arow1 = arow0 + (size_t)16 * K;
  const __bf16* brow0 = BT + (size_t)(n0 + 0  + l16) * K + kb0;
  const __bf16* brow1 = BT + (size_t)(n0 + 16 + l16) * K + kb0;
  const __bf16* brow2 = BT + (size_t)(n0 + 32 + l16) * K + kb0;
  const __bf16* brow3 = BT + (size_t)(n0 + 48 + l16) * K + kb0;

  v8f acc[2][4];
#pragma unroll
  for (int i = 0; i < 2; ++i)
#pragma unroll
    for (int j = 0; j < 4; ++j) acc[i][j] = (v8f){};

  // set X = current, set Y = prefetch
  V16U aX[2], bX[4], aY[2], bY[4];
  aX[0] = ldfragA(arow0, 0, ka0);
  aX[1] = ldfragA(arow1, 0, ka0);
  bX[0] = ldfragB(brow0);
  bX[1] = ldfragB(brow1);
  bX[2] = ldfragB(brow2);
  bX[3] = ldfragB(brow3);

  for (int kc = 0; kc < K; kc += 64) {
    // prefetch set Y @ kc+32 (always in range: K % 64 == 0)
    aY[0] = ldfragA(arow0, kc + 32, ka0);
    aY[1] = ldfragA(arow1, kc + 32, ka0);
    bY[0] = ldfragB(brow0 + kc + 32);
    bY[1] = ldfragB(brow1 + kc + 32);
    bY[2] = ldfragB(brow2 + kc + 32);
    bY[3] = ldfragB(brow3 + kc + 32);
    // compute set X
#pragma unroll
    for (int i = 0; i < 2; ++i)
#pragma unroll
      for (int j = 0; j < 4; ++j)
        acc[i][j] = __builtin_amdgcn_wmma_f32_16x16x32_bf16(
            false, aX[i].v, false, bX[j].v, (short)0, acc[i][j], false, false);
    // prefetch set X @ kc+64
    if (kc + 64 < K) {
      aX[0] = ldfragA(arow0, kc + 64, ka0);
      aX[1] = ldfragA(arow1, kc + 64, ka0);
      bX[0] = ldfragB(brow0 + kc + 64);
      bX[1] = ldfragB(brow1 + kc + 64);
      bX[2] = ldfragB(brow2 + kc + 64);
      bX[3] = ldfragB(brow3 + kc + 64);
    }
    // compute set Y
#pragma unroll
    for (int i = 0; i < 2; ++i)
#pragma unroll
      for (int j = 0; j < 4; ++j)
        acc[i][j] = __builtin_amdgcn_wmma_f32_16x16x32_bf16(
            false, aY[i].v, false, bY[j].v, (short)0, acc[i][j], false, false);
  }
  // C-frag layout: lane<16 -> rows r (VGPR r = M r); lane>=16 -> rows r+8
#pragma unroll
  for (int i = 0; i < 2; ++i)
#pragma unroll
    for (int j = 0; j < 4; ++j)
#pragma unroll
      for (int r = 0; r < 8; ++r) {
        int row = m0 + i * 16 + r + (hi ? 8 : 0);
        int col = n0 + j * 16 + l16;
        C[(size_t)row * N + col] = acc[i][j][r];
      }
}

// ---------------------------------------------------------------- RoPE
// lin: f32 [B*S, nh*HD]  ->  out: bf16 [B, nh, S, HD]
__global__ __launch_bounds__(256) void k_rope(const float* __restrict__ lin,
                                              __bf16* __restrict__ out, int nh) {
  int i = blockIdx.x * blockDim.x + threadIdx.x;
  int total = B_ * nh * S_ * 64;
  if (i >= total) return;
  int j = i & 63;
  int s = (i >> 6) & (S_ - 1);
  int h = (i >> 17) % nh;         // 64*S_ = 2^17
  int b = i / (131072 * nh);
  float invf = __expf(-(float)j * (9.210340371976184f / 64.0f)); // 10000^(-j/64)
  float ang = (float)s * invf;
  float c, sn;
  __sincosf(ang, &sn, &c);
  const float* row = lin + (size_t)(b * S_ + s) * (nh * HD_) + h * HD_;
  float x1 = row[j], x2 = row[j + 64];
  __bf16* orow = out + ((size_t)(b * nh + h) * S_ + s) * HD_;
  orow[j]      = (__bf16)(x1 * c - x2 * sn);
  orow[j + 64] = (__bf16)(x2 * c + x1 * sn);
}

// vlin: f32 [B*S, KV*HD]  ->  vt: bf16 [B, KV, HD, S]  (transposed for PV B-frags)
__global__ __launch_bounds__(256) void k_vtrans(const float* __restrict__ vlin,
                                                __bf16* __restrict__ vt) {
  int i = blockIdx.x * blockDim.x + threadIdx.x;
  if (i >= B_ * S_ * KV_ * HD_) return;
  int hd = i % HD_;
  int kv = (i / HD_) % KV_;
  int s  = (i / (HD_ * KV_)) % S_;
  int b  = i / (HD_ * KV_ * S_);
  vt[((size_t)(b * KV_ + kv) * HD_ + hd) * S_ + s] = (__bf16)vlin[i];
}

// ---------------------------------------------------------------- Flash attention
// One wave (32-thread block) per 16-query tile per (b,h). Online softmax,
// causal, GQA (kvh = h/4). 32 keys per iteration: 8 WMMAs for QK^T, 8 for PV.
__global__ __launch_bounds__(32) void k_attn(const __bf16* __restrict__ q,   // [B,H,S,HD]
                                             const __bf16* __restrict__ k,   // [B,KV,S,HD]
                                             const __bf16* __restrict__ vt,  // [B,KV,HD,S]
                                             __bf16* __restrict__ attn) {    // [B*S, H*HD]
  __shared__ __bf16 lds_p[16 * 32];
  const int lane = threadIdx.x;
  const int qt = blockIdx.x & 127;        // S_/16
  const int h  = (blockIdx.x >> 7) & 15;
  const int b  = blockIdx.x >> 11;
  const int kvh = h >> 2;
  const int qbase = qt << 4;
  const int l16 = lane & 15;
  const int hi  = lane >> 4;
  const int ka0 = hi ? 8 : 0;
  const int koff = hi ? 8 : 0;

  const __bf16* qh = q  + ((size_t)(b * H_  + h)   * S_)  * HD_;
  const __bf16* kh = k  + ((size_t)(b * KV_ + kvh) * S_)  * HD_;
  const __bf16* vh = vt + ((size_t)(b * KV_ + kvh) * HD_) * S_;

  // Q A-fragments for the 4 K-chunks of HD=128 (held in registers for all key blocks)
  V16U qa[4];
  const __bf16* qrow = qh + (size_t)(qbase + l16) * HD_;
#pragma unroll
  for (int c = 0; c < 4; ++c) {
    qa[c].h8[0] = *(const v8bf*)(qrow + 32 * c + ka0);
    qa[c].h8[1] = *(const v8bf*)(qrow + 32 * c + ka0 + 16);
  }

  v8f oacc[8];
#pragma unroll
  for (int t = 0; t < 8; ++t) oacc[t] = (v8f){};
  float mrun[8], lrun[8];
#pragma unroll
  for (int r = 0; r < 8; ++r) { mrun[r] = -1e30f; lrun[r] = 0.0f; }

  for (int kb = 0; kb < qbase + 16; kb += 32) {
    // ---- scores: S = Q * K^T (two 16x16 N-tiles)
    v8f s[2];
#pragma unroll
    for (int ntile = 0; ntile < 2; ++ntile) {
      v8f acc = {};
      const __bf16* krow = kh + (size_t)(kb + ntile * 16 + l16) * HD_ + koff;
#pragma unroll
      for (int c = 0; c < 4; ++c) {
        V16U kf;
        kf.h8[0] = *(const v8bf*)(krow + 32 * c);
        kf.h8[1] = *(const v8bf*)(krow + 32 * c + 16);
        acc = __builtin_amdgcn_wmma_f32_16x16x32_bf16(
            false, qa[c].v, false, kf.v, (short)0, acc, false, false);
      }
      s[ntile] = acc;
    }
    // ---- scale + causal mask (C-frag: row = r + 8*hi, col = lane%16)
#pragma unroll
    for (int ntile = 0; ntile < 2; ++ntile) {
      int colg = kb + ntile * 16 + l16;
#pragma unroll
      for (int r = 0; r < 8; ++r) {
        int rowg = qbase + r + (hi ? 8 : 0);
        float sv = s[ntile][r] * SCALE_;
        s[ntile][r] = (colg > rowg) ? -1e30f : sv;
      }
    }
    __syncthreads();  // previous iteration's LDS reads are done before rewrite
    // ---- online softmax (row reductions across the 16-lane halves)
#pragma unroll
    for (int r = 0; r < 8; ++r) {
      float bm = fmaxf(s[0][r], s[1][r]);
      bm = fmaxf(bm, __shfl_xor(bm, 1, 32));
      bm = fmaxf(bm, __shfl_xor(bm, 2, 32));
      bm = fmaxf(bm, __shfl_xor(bm, 4, 32));
      bm = fmaxf(bm, __shfl_xor(bm, 8, 32));
      float mnew = fmaxf(mrun[r], bm);
      float corr = __expf(mrun[r] - mnew);
      float p0 = __expf(s[0][r] - mnew);
      float p1 = __expf(s[1][r] - mnew);
      float ps = p0 + p1;
      ps += __shfl_xor(ps, 1, 32);
      ps += __shfl_xor(ps, 2, 32);
      ps += __shfl_xor(ps, 4, 32);
      ps += __shfl_xor(ps, 8, 32);
      lrun[r] = lrun[r] * corr + ps;
      mrun[r] = mnew;
      int mrow = r + (hi ? 8 : 0);
      lds_p[mrow * 32 + l16]      = (__bf16)p0;
      lds_p[mrow * 32 + 16 + l16] = (__bf16)p1;
#pragma unroll
      for (int t = 0; t < 8; ++t) oacc[t][r] *= corr;
    }
    __syncthreads();
    // ---- P as A-fragment (16x32) from LDS
    V16U pf;
    pf.h8[0] = *(const v8bf*)(&lds_p[l16 * 32 + ka0]);
    pf.h8[1] = *(const v8bf*)(&lds_p[l16 * 32 + ka0 + 16]);
    // ---- O += P * V   (8 hd-tiles of 16)
#pragma unroll
    for (int t = 0; t < 8; ++t) {
      const __bf16* vcol = vh + (size_t)(t * 16 + l16) * S_ + kb + koff;
      V16U vf;
      vf.h8[0] = *(const v8bf*)(vcol);
      vf.h8[1] = *(const v8bf*)(vcol + 16);
      oacc[t] = __builtin_amdgcn_wmma_f32_16x16x32_bf16(
          false, pf.v, false, vf.v, (short)0, oacc[t], false, false);
    }
  }
  // ---- epilogue: normalize and store to [B*S, H*HD] bf16
#pragma unroll
  for (int r = 0; r < 8; ++r) {
    float invl = 1.0f / lrun[r];
    int rowg = qbase + r + (hi ? 8 : 0);
    __bf16* orow = attn + ((size_t)(b * S_) + rowg) * (H_ * HD_) + h * HD_;
#pragma unroll
    for (int t = 0; t < 8; ++t)
      orow[t * 16 + l16] = (__bf16)(oacc[t][r] * invl);
  }
}

// ---------------------------------------------------------------- launch
extern "C" void kernel_launch(void* const* d_in, const int* in_sizes, int n_in,
                              void* d_out, int out_size, void* d_ws, size_t ws_size,
                              hipStream_t stream) {
  (void)in_sizes; (void)n_in; (void)out_size; (void)ws_size;
  const float* x  = (const float*)d_in[0];
  const float* wq = (const float*)d_in[1];
  const float* wk = (const float*)d_in[2];
  const float* wv = (const float*)d_in[3];
  const float* wo = (const float*)d_in[4];
  float* out = (float*)d_out;

  char* p = (char*)d_ws;
  auto carve = [&](size_t bytes) {
    char* r = p;
    p += (bytes + 255) & ~(size_t)255;
    return (void*)r;
  };
  __bf16* xb   = (__bf16*)carve((size_t)4096 * 2048 * 2);
  __bf16* wqT  = (__bf16*)carve((size_t)2048 * 2048 * 2);
  __bf16* wkT  = (__bf16*)carve((size_t)512  * 2048 * 2);
  __bf16* wvT  = (__bf16*)carve((size_t)512  * 2048 * 2);
  __bf16* woT  = (__bf16*)carve((size_t)2048 * 2048 * 2);
  float*  qlin = (float*) carve((size_t)4096 * 2048 * 4);
  float*  klin = (float*) carve((size_t)4096 * 512  * 4);
  float*  vlin = (float*) carve((size_t)4096 * 512  * 4);
  __bf16* qro  = (__bf16*)carve((size_t)B_ * H_  * S_ * HD_ * 2);
  __bf16* kro  = (__bf16*)carve((size_t)B_ * KV_ * S_ * HD_ * 2);
  __bf16* vT   = (__bf16*)carve((size_t)B_ * KV_ * HD_ * S_ * 2);
  __bf16* attn = (__bf16*)qlin;  // qlin is dead after RoPE; reuse its region

  // 1) precision/layout prep
  k_cast_bf16<<<8388608 / 256, 256, 0, stream>>>(x, xb, 8388608);
  k_transpose_bf16<<<4194304 / 256, 256, 0, stream>>>(wq, wqT, 2048, 2048);
  k_transpose_bf16<<<1048576 / 256, 256, 0, stream>>>(wk, wkT, 2048, 512);
  k_transpose_bf16<<<1048576 / 256, 256, 0, stream>>>(wv, wvT, 2048, 512);
  k_transpose_bf16<<<4194304 / 256, 256, 0, stream>>>(wo, woT, 2048, 2048);
  // 2) QKV projections (WMMA, 32x64 wave tiles, double-buffered)
  k_gemm_bf16<<<512, 256, 0, stream>>>(xb, wqT, qlin, 4096, 2048, 2048);
  k_gemm_bf16<<<128, 256, 0, stream>>>(xb, wkT, klin, 4096, 512, 2048);
  k_gemm_bf16<<<128, 256, 0, stream>>>(xb, wvT, vlin, 4096, 512, 2048);
  // 3) RoPE + layout transforms
  k_rope<<<4194304 / 256, 256, 0, stream>>>(qlin, qro, 16);
  k_rope<<<1048576 / 256, 256, 0, stream>>>(klin, kro, 4);
  k_vtrans<<<2097152 / 256, 256, 0, stream>>>(vlin, vT);
  // 4) flash attention (WMMA)
  k_attn<<<B_ * H_ * (S_ / 16), 32, 0, stream>>>(qro, kro, vT, attn);
  // 5) output projection (WMMA)
  k_gemm_bf16<<<512, 256, 0, stream>>>(attn, woT, out, 4096, 2048, 2048);
}
